// UltrametricCausalSelfAttention_74096775790958
// MI455X (gfx1250) — matrix operand order repaired and checked by reading, not compiled, not run on hardware
//
#include <hip/hip_runtime.h>
#include <hip/hip_bf16.h>

#define T_SEQ 2048
#define C_EMB 2048
#define NHEAD 16
#define HDIM  128

typedef __attribute__((ext_vector_type(16))) __bf16 bf16x16;
typedef __attribute__((ext_vector_type(8)))  float  floatx8;

union AFrag { bf16x16 v; unsigned int u[8]; };

__device__ __forceinline__ unsigned short f2bf(float f) {
  unsigned int u = __float_as_uint(f);
  u += 0x7FFFu + ((u >> 16) & 1u);          // round-to-nearest-even
  return (unsigned short)(u >> 16);
}

// A-fragment K offset for 16-bit 16x32 A matrix (CDNA5 layout):
// VGPR i<4: K = 2i + 8*half ; i>=4: K = 16 + 2(i-4) + 8*half
__device__ __forceinline__ int a_k0(int i, int h) {
  return (i < 4 ? 2 * i : 16 + 2 * (i - 4)) + 8 * h;
}

// Async DMA: global -> LDS, 16 bytes per lane, tracked by ASYNCcnt.
// LDS address = low 32 bits of the generic pointer (ISA 10.2: flat LDS
// addresses truncate to a 32-bit offset from the workgroup LDS base).
__device__ __forceinline__ void async_b128(void* lds_dst, const void* gsrc) {
  unsigned ldsoff = (unsigned)(size_t)lds_dst;
  unsigned long long ga = (unsigned long long)(size_t)gsrc;
  asm volatile("global_load_async_to_lds_b128 %0, %1, off"
               :: "v"(ldsoff), "v"(ga) : "memory");
}
__device__ __forceinline__ void wait_async0() {
  asm volatile("s_wait_asynccnt 0x0" ::: "memory");
}

// ---------------------------------------------------------------- f32 -> bf16
__global__ void cvt_bf16(const float* __restrict__ in,
                         unsigned short* __restrict__ out, int n) {
  int idx = (blockIdx.x * blockDim.x + threadIdx.x) * 4;
  if (idx >= n) return;
  float4 f = *(const float4*)(in + idx);
  unsigned int lo = (unsigned int)f2bf(f.x) | ((unsigned int)f2bf(f.y) << 16);
  unsigned int hi = (unsigned int)f2bf(f.z) | ((unsigned int)f2bf(f.w) << 16);
  *(uint2*)(out + idx) = make_uint2(lo, hi);
}

// -------------------------------------------- Out[M,N] = A[M,K] @ W[N,K]^T
// 128x128 tile per block, 8 waves, wave w owns rows [16w,16w+16), 8 n-tiles.
// Double-buffered LDS, async global->LDS DMA overlapped with WMMA.
__global__ __launch_bounds__(256) void gemm_xwT_bf16(
    const unsigned short* __restrict__ A, const unsigned short* __restrict__ W,
    float* __restrict__ Out, int M, int N, int K) {
  __shared__ __align__(16) unsigned short lds_a[2][128 * 32];
  __shared__ __align__(16) unsigned short lds_w[2][128 * 32];
  const int tid = threadIdx.x;
  const int lane = tid & 31;
  const int wv = tid >> 5;
  const int half = lane >> 4;
  const int l16 = lane & 15;
  const int m0 = blockIdx.y * 128;
  const int n0 = blockIdx.x * 128;
  floatx8 acc[8] = {};

  auto stage = [&](int buf, int kk) {
#pragma unroll
    for (int r = 0; r < 2; ++r) {              // 2x 8KB tiles, 16B per lane
      int c = tid + 256 * r;
      int row = c >> 2, qo = c & 3;
      async_b128(&((uint4*)lds_a[buf])[c],
                 (const uint4*)(A + (size_t)(m0 + row) * K + kk) + qo);
      async_b128(&((uint4*)lds_w[buf])[c],
                 (const uint4*)(W + (size_t)(n0 + row) * K + kk) + qo);
    }
  };

  stage(0, 0);
  wait_async0();
  __syncthreads();

  int buf = 0;
  for (int kk = 0; kk < K; kk += 32) {
    if (kk + 32 < K) stage(buf ^ 1, kk + 32);  // DMA next tile during compute

    AFrag af;
    const int arow = 16 * wv + l16;
#pragma unroll
    for (int i = 0; i < 8; ++i)
      af.u[i] = *(const unsigned int*)&lds_a[buf][arow * 32 + a_k0(i, half)];
#pragma unroll
    for (int j = 0; j < 8; ++j) {
      AFrag bw;
#pragma unroll
      for (int i = 0; i < 8; ++i)   // B: col = lane%16, K pairs a_k0(i, half)
        bw.u[i] = *(const unsigned int*)&lds_w[buf][(16 * j + l16) * 32 + a_k0(i, half)];
      acc[j] = __builtin_amdgcn_wmma_f32_16x16x32_bf16(
          false, af.v, false, bw.v, (short)0, acc[j], false, false);
    }

    wait_async0();
    __syncthreads();
    buf ^= 1;
  }

#pragma unroll
  for (int j = 0; j < 8; ++j)
#pragma unroll
    for (int i = 0; i < 8; ++i) {   // C layout: row = i + 8*half, col = lane%16
      int row = m0 + 16 * wv + i + 8 * half;
      int col = n0 + 16 * j + l16;
      Out[(size_t)row * N + col] = acc[j][i];
    }
}

// -------------------- fused RoPE + RMSNorm + ultrametric 7-bit code, 1 wave/(h,t)
__global__ __launch_bounds__(256) void rope_norm_code(
    const float* __restrict__ src, const float* __restrict__ cs,
    const float* __restrict__ sn, const float* __restrict__ Wd,
    unsigned short* __restrict__ outn, int* __restrict__ codes) {
  const int lane = threadIdx.x & 31;
  const int wv = threadIdx.x >> 5;
  const int pair = blockIdx.x * 8 + wv;
  const int t = pair & (T_SEQ - 1);
  const int h = pair >> 11;
  const float* base = src + (size_t)t * C_EMB + h * HDIM;
  const int j0 = lane, j1 = lane + 32;
  float x1a = base[j0], x2a = base[64 + j0];
  float x1b = base[j1], x2b = base[64 + j1];
  float ca = cs[t * 64 + j0], sa = sn[t * 64 + j0];
  float cb = cs[t * 64 + j1], sb = sn[t * 64 + j1];
  float r1a =  x1a * ca + x2a * sa;
  float r2a = -x1a * sa + x2a * ca;
  float r1b =  x1b * cb + x2b * sb;
  float r2b = -x1b * sb + x2b * cb;
  float ss = r1a * r1a + r2a * r2a + r1b * r1b + r2b * r2b;
#pragma unroll
  for (int off = 1; off < 32; off <<= 1) ss += __shfl_xor(ss, off, 32);
  float scale = rsqrtf(ss * (1.0f / 128.0f) + 1e-6f);
  float v0 = r1a * scale;   // d = lane
  float v1 = r1b * scale;   // d = lane+32
  float v2 = r2a * scale;   // d = 64+lane
  float v3 = r2b * scale;   // d = 96+lane
  unsigned short* od = outn + (size_t)h * T_SEQ * HDIM + (size_t)t * HDIM;
  od[j0] = f2bf(v0);  od[j1] = f2bf(v1);
  od[64 + j0] = f2bf(v2);  od[64 + j1] = f2bf(v3);
  int code = 0;
#pragma unroll
  for (int kk = 0; kk < 7; ++kk) {   // first K-1 digits, P=2 -> bits
    const float* wr = Wd + kk * HDIM;
    float z = v0 * wr[lane] + v1 * wr[lane + 32] + v2 * wr[lane + 64] + v3 * wr[lane + 96];
#pragma unroll
    for (int off = 1; off < 32; off <<= 1) z += __shfl_xor(z, off, 32);
    float sg = 1.0f / (1.0f + __expf(-z));
    code |= ((sg * 1.99f >= 1.0f) ? 1 : 0) << kk;
  }
  if (lane == 0) codes[h * T_SEQ + t] = code;
}

// --------------------------------- v[T, H*D] f32  ->  vt[H][D][T] bf16
__global__ void relayout_vt(const float* __restrict__ vf,
                            unsigned short* __restrict__ vt) {
  size_t idx = (size_t)blockIdx.x * blockDim.x + threadIdx.x;
  int t = (int)(idx & (T_SEQ - 1));
  int rest = (int)(idx >> 11);
  int d = rest & (HDIM - 1);
  int h = rest >> 7;
  vt[idx] = f2bf(vf[(size_t)t * C_EMB + h * HDIM + d]);
}

// ---------------- flash-style ultrametric-masked causal attention, 1 block = (h, 128 q)
__global__ __launch_bounds__(256) void attn_kernel(
    const unsigned short* __restrict__ qn, const unsigned short* __restrict__ kn,
    const unsigned short* __restrict__ vt, const int* __restrict__ qcodes,
    const int* __restrict__ kcodes, unsigned short* __restrict__ yb) {
  __shared__ __align__(16) unsigned short lds_k[2][32 * 128];   // [key][d]
  __shared__ __align__(16) unsigned short lds_v[2][128 * 32];   // [d][key]
  __shared__ int lds_kc[2][32];
  __shared__ __align__(16) unsigned short lds_p[8][16 * 32];    // per-wave C->A scratch
  const int tid = threadIdx.x, lane = tid & 31, wv = tid >> 5;
  const int half = lane >> 4, l16 = lane & 15;
  const int h = blockIdx.y;
  const int q0 = blockIdx.x * 128;
  const int qrow0 = q0 + 16 * wv;
  const unsigned short* qn_h = qn + (size_t)h * T_SEQ * HDIM;
  const unsigned short* kn_h = kn + (size_t)h * T_SEQ * HDIM;
  const unsigned short* vt_h = vt + (size_t)h * HDIM * T_SEQ;
  const int* qc_h = qcodes + h * T_SEQ;
  const int* kc_h = kcodes + h * T_SEQ;

  auto stage = [&](int buf, int kt) {
#pragma unroll
    for (int r = 0; r < 2; ++r) {              // K tile (contig) and V^T tile
      int c = tid + 256 * r;
      async_b128(&((uint4*)lds_k[buf])[c], ((const uint4*)kn_h) + (size_t)kt * 16 + c);
      int d = c >> 2, cc = c & 3;
      async_b128(&((uint4*)lds_v[buf])[c], (const uint4*)(vt_h + (size_t)d * T_SEQ + kt) + cc);
    }
    if (tid < 32) lds_kc[buf][tid] = kc_h[kt + tid];
  };

  AFrag qa[4];                                   // q A-frags, K = d (4 chunks of 32)
#pragma unroll
  for (int c = 0; c < 4; ++c)
#pragma unroll
    for (int i = 0; i < 8; ++i)
      qa[c].u[i] = *(const unsigned int*)&qn_h[(size_t)(qrow0 + l16) * HDIM + 32 * c + a_k0(i, half)];

  int qcv[8];
#pragma unroll
  for (int i = 0; i < 8; ++i) qcv[i] = qc_h[qrow0 + i + 8 * half];

  floatx8 yacc[8] = {};
  float mrow[8], lrow[8];
#pragma unroll
  for (int i = 0; i < 8; ++i) { mrow[i] = -3.0e38f; lrow[i] = 0.f; }

  const float sscale = 0.08838834764831845f;     // 1/sqrt(128)
  const int kend = q0 + 128;

  stage(0, 0);
  wait_async0();
  __syncthreads();

  int buf = 0;
  for (int kt = 0; kt < kend; kt += 32) {
    if (kt + 32 < kend) stage(buf ^ 1, kt + 32);   // DMA next tile during compute

    floatx8 s0 = {}, s1 = {};
#pragma unroll
    for (int c = 0; c < 4; ++c) {                // S = q . k^T over d
      AFrag b0, b1;
#pragma unroll
      for (int i = 0; i < 8; ++i) {
        int ko = 32 * c + a_k0(i, half);
        b0.u[i] = *(const unsigned int*)&lds_k[buf][l16 * 128 + ko];
        b1.u[i] = *(const unsigned int*)&lds_k[buf][(16 + l16) * 128 + ko];
      }
      s0 = __builtin_amdgcn_wmma_f32_16x16x32_bf16(false, qa[c].v, false, b0.v, (short)0, s0, false, false);
      s1 = __builtin_amdgcn_wmma_f32_16x16x32_bf16(false, qa[c].v, false, b1.v, (short)0, s1, false, false);
    }

    const int kc0 = lds_kc[buf][l16], kc1 = lds_kc[buf][16 + l16];
    const int key0 = kt + l16, key1 = kt + 16 + l16;
    float p0[8], p1[8];
#pragma unroll
    for (int i = 0; i < 8; ++i) {                // online softmax in C layout
      int row = qrow0 + i + 8 * half;
      float sv0 = s0[i] * sscale;
      float sv1 = s1[i] * sscale;
      if (!(key0 <= row && kc0 == qcv[i])) sv0 = -1.0e30f;
      if (!(key1 <= row && kc1 == qcv[i])) sv1 = -1.0e30f;
      float rm = fmaxf(sv0, sv1);
#pragma unroll
      for (int off = 1; off < 16; off <<= 1) rm = fmaxf(rm, __shfl_xor(rm, off, 32));
      float mn = fmaxf(mrow[i], rm);
      float alpha = __expf(mrow[i] - mn);
      float e0 = (sv0 > -5.0e29f) ? __expf(sv0 - mn) : 0.f;
      float e1 = (sv1 > -5.0e29f) ? __expf(sv1 - mn) : 0.f;
      float rs = e0 + e1;
#pragma unroll
      for (int off = 1; off < 16; off <<= 1) rs += __shfl_xor(rs, off, 32);
      lrow[i] = lrow[i] * alpha + rs;
      mrow[i] = mn;
#pragma unroll
      for (int j = 0; j < 8; ++j) yacc[j][i] *= alpha;
      p0[i] = e0; p1[i] = e1;
    }

    unsigned short* pl = lds_p[wv];              // C-layout -> A-layout via LDS
#pragma unroll
    for (int i = 0; i < 8; ++i) {
      int prow = i + 8 * half;
      pl[prow * 32 + l16] = f2bf(p0[i]);
      pl[prow * 32 + 16 + l16] = f2bf(p1[i]);
    }
    asm volatile("s_wait_dscnt 0" ::: "memory"); // wave-private scratch, LDS in-order
    AFrag pa;
#pragma unroll
    for (int i = 0; i < 8; ++i)
      pa.u[i] = *(const unsigned int*)&pl[l16 * 32 + a_k0(i, half)];
#pragma unroll
    for (int j = 0; j < 8; ++j) {                // y += P . V
      AFrag bv;
#pragma unroll
      for (int i = 0; i < 8; ++i)
        bv.u[i] = *(const unsigned int*)&lds_v[buf][(16 * j + l16) * 32 + a_k0(i, half)];
      yacc[j] = __builtin_amdgcn_wmma_f32_16x16x32_bf16(false, pa.v, false, bv.v, (short)0, yacc[j], false, false);
    }

    wait_async0();
    __syncthreads();
    buf ^= 1;
  }

  float invl[8];
#pragma unroll
  for (int i = 0; i < 8; ++i) invl[i] = (lrow[i] > 0.f) ? 1.0f / lrow[i] : 0.f;
#pragma unroll
  for (int j = 0; j < 8; ++j)
#pragma unroll
    for (int i = 0; i < 8; ++i) {
      int row = qrow0 + i + 8 * half;
      int col = h * HDIM + 16 * j + l16;
      yb[(size_t)row * C_EMB + col] = f2bf(yacc[j][i] * invl[i]);
    }
}

// --------------------------------------------------------------------------
extern "C" void kernel_launch(void* const* d_in, const int* in_sizes, int n_in,
                              void* d_out, int out_size, void* d_ws, size_t ws_size,
                              hipStream_t stream) {
  (void)in_sizes; (void)n_in; (void)out_size; (void)ws_size;
  const float* x    = (const float*)d_in[0];
  const float* cosb = (const float*)d_in[1];
  const float* sinb = (const float*)d_in[2];
  const float* Wq   = (const float*)d_in[3];
  const float* Wk   = (const float*)d_in[4];
  const float* Wv   = (const float*)d_in[5];
  const float* Wpj  = (const float*)d_in[6];
  const float* Wdq  = (const float*)d_in[7];
  const float* Wdk  = (const float*)d_in[8];
  float* out = (float*)d_out;

  char* ws = (char*)d_ws;
  const size_t MB = 1024ull * 1024ull;
  unsigned short* xb  = (unsigned short*)(ws + 0);
  unsigned short* wqb = (unsigned short*)(ws + 8 * MB);
  unsigned short* wkb = (unsigned short*)(ws + 16 * MB);
  unsigned short* wvb = (unsigned short*)(ws + 24 * MB);
  unsigned short* wpb = (unsigned short*)(ws + 32 * MB);
  float* qf = (float*)(ws + 40 * MB);
  float* kf = (float*)(ws + 56 * MB);
  float* vf = (float*)(ws + 72 * MB);
  unsigned short* qn = (unsigned short*)(ws + 88 * MB);
  unsigned short* kn = (unsigned short*)(ws + 96 * MB);
  unsigned short* vt = (unsigned short*)(ws + 104 * MB);
  unsigned short* yb = (unsigned short*)(ws + 112 * MB);
  int* qc = (int*)(ws + 120 * MB);
  int* kc = (int*)(ws + 120 * MB + 256 * 1024);

  const int NEL = T_SEQ * C_EMB;
  dim3 cb(256), cg(NEL / (256 * 4));
  cvt_bf16<<<cg, cb, 0, stream>>>(x,   xb,  NEL);
  cvt_bf16<<<cg, cb, 0, stream>>>(Wq,  wqb, NEL);
  cvt_bf16<<<cg, cb, 0, stream>>>(Wk,  wkb, NEL);
  cvt_bf16<<<cg, cb, 0, stream>>>(Wv,  wvb, NEL);
  cvt_bf16<<<cg, cb, 0, stream>>>(Wpj, wpb, NEL);

  dim3 gg(16, 16), gb(256);
  gemm_xwT_bf16<<<gg, gb, 0, stream>>>(xb, wqb, qf, T_SEQ, C_EMB, C_EMB);
  gemm_xwT_bf16<<<gg, gb, 0, stream>>>(xb, wkb, kf, T_SEQ, C_EMB, C_EMB);
  gemm_xwT_bf16<<<gg, gb, 0, stream>>>(xb, wvb, vf, T_SEQ, C_EMB, C_EMB);

  rope_norm_code<<<dim3(4096), dim3(256), 0, stream>>>(qf, cosb, sinb, Wdq, qn, qc);
  rope_norm_code<<<dim3(4096), dim3(256), 0, stream>>>(kf, cosb, sinb, Wdk, kn, kc);
  relayout_vt<<<dim3(NEL / 256), dim3(256), 0, stream>>>(vf, vt);

  attn_kernel<<<dim3(T_SEQ / 128, NHEAD), dim3(256), 0, stream>>>(qn, kn, vt, qc, kc, yb);

  gemm_xwT_bf16<<<gg, gb, 0, stream>>>(yb, wpb, out, T_SEQ, C_EMB, C_EMB);
}
